// DCAM_62904091018232
// MI455X (gfx1250) — compile-verified, matrix operand order checked by reading, not run on hardware
//
#include <hip/hip_runtime.h>

// ---------------------------------------------------------------------------
// DCAM attention for MI455X (gfx1250, wave32, WMMA).
//   B=8, C=256, M=32, H=W=128, DS=2 -> pooled 64x64, N=4096.
// Roofline: 82 GFLOP vs 540 MB mandatory HBM traffic @23.3TB/s -> keep the
// 4096x4096 attention matrix out of memory (flash attention), run all matmuls
// on v_wmma_f32_16x16x32_bf16 with f32 accumulation, fuse the 2x2 avg-pool
// into the projection GEMM loads, and feed the hot P*V^T loop from LDS
// (V slab staged once per workgroup via async global->LDS, shared by 4 waves).
// ---------------------------------------------------------------------------

typedef __bf16 bf16;
typedef __attribute__((ext_vector_type(16))) __bf16 v16bf;
typedef __attribute__((ext_vector_type(8)))  __bf16 bf16x8;
typedef __attribute__((ext_vector_type(8)))  float  v8f;
typedef __attribute__((ext_vector_type(4)))  int    v4i;

#define BSZ   8
#define CDIM  256
#define MQ    32
#define HF    128
#define WF    128
#define HP    64
#define NP    4096   // HP*HP pooled positions
#define VPAD  40     // padded LDS row: 80B (16B-aligned vec reads, 16 distinct banks)

// gfx1250 async global->LDS path (ASYNCcnt-tracked); falls back to plain
// load + ds_store if the builtins are not declared by this toolchain.
#if defined(__has_builtin)
#if __has_builtin(__builtin_amdgcn_global_load_async_to_lds_b128) && \
    __has_builtin(__builtin_amdgcn_s_wait_asynccnt)
#define USE_ASYNC_LDS 1
#endif
#endif
#ifndef USE_ASYNC_LDS
#define USE_ASYNC_LDS 0
#endif

static __device__ __forceinline__ v16bf concat8(bf16x8 lo, bf16x8 hi) {
    return __builtin_shufflevector(lo, hi, 0, 1, 2, 3, 4, 5, 6, 7,
                                           8, 9, 10, 11, 12, 13, 14, 15);
}

// ---------------------------------------------------------------------------
// Kernel 1: fused 2x2 avg-pool + 1x1-conv projection GEMM.
//   Out(Mdim x NP) = W(Mdim x 256) @ pooled(256 x NP)   per batch.
// Block = 128 threads (4 waves); block tile = 16 (M) x 64 (N); K step = 32.
// outT=1 stores position-major (B,N,Mdim) for q/k; outT=0 stores (B,Mdim,N).
// ---------------------------------------------------------------------------
__global__ __launch_bounds__(128)
void pool_proj_gemm(const float* __restrict__ Wmat, const float* __restrict__ F,
                    bf16* __restrict__ Out, int Mdim, int outT)
{
    __shared__ bf16 Xs[32][64];               // 4 KB K x N tile (pooled, bf16)
    const int b    = blockIdx.z;
    const int m0   = blockIdx.y * 16;
    const int n0   = blockIdx.x * 64;         // one full pooled row: h'=n0>>6
    const int t    = threadIdx.x;
    const int wv   = t >> 5;
    const int ln   = t & 31;
    const int l16  = ln & 15;
    const int hl   = ln >> 4;                 // half-wave id (0/1)

    const int hp = n0 >> 6;                   // pooled row
    const float* Fb = F + (size_t)b * CDIM * (HF * WF);

    v8f acc = {};
    for (int kk = 0; kk < CDIM; kk += 32) {
        // ---- stage pooled tile: Xs[kr][col] = avg2x2(F[b, kk+kr, 2hp, 2col])
        #pragma unroll
        for (int i = 0; i < 16; ++i) {
            int linear = i * 128 + t;         // 0..2047
            int kr  = linear >> 6;            // 0..31
            int col = linear & 63;            // pooled w'
            const float* fp = Fb + ((size_t)(kk + kr) * HF + 2 * hp) * WF + 2 * col;
            float v = 0.25f * (fp[0] + fp[1] + fp[WF] + fp[WF + 1]);
            Xs[kr][col] = (bf16)v;
        }
        __syncthreads();

        // ---- A-regs (W tile 16x32): lane row m0+l16; K = hl*8+e / 16+hl*8+e
        v16bf a;
        {
            const float* wp = Wmat + (size_t)(m0 + l16) * CDIM + kk + hl * 8;
            #pragma unroll
            for (int e = 0; e < 8; ++e) { a[e] = (bf16)wp[e]; a[8 + e] = (bf16)wp[16 + e]; }
        }
        // ---- B-regs from LDS: lane col = wv*16+l16; K per CDNA5 B layout
        v16bf bb;
        {
            int col = wv * 16 + l16;
            #pragma unroll
            for (int j = 0; j < 16; ++j) {
                int kr = ((j < 8) ? 0 : 16) + hl * 8 + (j & 7);
                bb[j] = Xs[kr][col];
            }
        }
        acc = __builtin_amdgcn_wmma_f32_16x16x32_bf16(false, a, false, bb,
                                                      (short)0, acc, false, false);
        __syncthreads();
    }

    // ---- store (D layout: lane col n, VGPR r -> row m0 + r + 8*hl)
    const int n = n0 + wv * 16 + l16;
    if (outT) {
        bf16* op = Out + ((size_t)b * NP + n) * Mdim + m0 + 8 * hl;
        #pragma unroll
        for (int r = 0; r < 8; ++r) op[r] = (bf16)acc[r];
    } else {
        #pragma unroll
        for (int r = 0; r < 8; ++r) {
            int m = m0 + r + 8 * hl;
            Out[((size_t)b * Mdim + m) * NP + n] = (bf16)acc[r];
        }
    }
}

// ---------------------------------------------------------------------------
// Kernel 2: flash attention.
//   S(16x32) = Qtile(16x32d) x Ktile        -> 2 WMMA (d=32 = one K-step)
//   online softmax (row stats on 16-lane halves via shfl_xor)
//   O(16x256) += P(16x32) x V^T             -> 16 WMMA per key block
// Block = 4 waves / 64 query rows. The 32-key x 256-ch V slab (16 KB) is
// staged into LDS once per block (async global->LDS when available) and
// shared by all 4 waves; P transposes D->A layout through padded LDS.
// ---------------------------------------------------------------------------
__global__ __launch_bounds__(128)
void flash_attn_kernel(const bf16* __restrict__ qT, const bf16* __restrict__ kT,
                       const bf16* __restrict__ Vb, float* __restrict__ O)
{
    __shared__ __align__(16) bf16 Vs[CDIM][VPAD];   // 20 KB V slab (keys contig)
    __shared__ __align__(16) bf16 Ps[4][16][VPAD];  // 5 KB per-wave P staging
    const int b   = blockIdx.y;
    const int t   = threadIdx.x;
    const int wv  = t >> 5;
    const int ln  = t & 31;
    const int l16 = ln & 15;
    const int hl  = ln >> 4;
    const int i0  = blockIdx.x * 64 + wv * 16;      // this wave's query rows

    // Q A-regs: row i0+l16, contiguous 32 dims (position-major layout)
    v16bf aq;
    {
        const bf16* qp = qT + ((size_t)b * NP + i0 + l16) * MQ + hl * 8;
        aq = concat8(*(const bf16x8*)qp, *(const bf16x8*)(qp + 16));
    }

    float acc[16][8];                               // 16 channel tiles x 8 rows
    #pragma unroll
    for (int ct = 0; ct < 16; ++ct)
        #pragma unroll
        for (int r = 0; r < 8; ++r) acc[ct][r] = 0.0f;
    float rm[8], rs[8];
    #pragma unroll
    for (int r = 0; r < 8; ++r) { rm[r] = -1e30f; rs[r] = 0.0f; }

    const bf16* kTb = kT + (size_t)b * NP * MQ;
    const bf16* Vbb = Vb + (size_t)b * CDIM * NP;

    // staging coordinates: 4 lanes cover one 64B V row (32 keys), 8 passes
    const int srow = t >> 2;                        // 0..31 per pass
    const int spart = t & 3;                        // 16B chunk within row

    for (int j0 = 0; j0 < NP; j0 += 32) {
        // ---- stage V slab: Vs[c][kk] = Vb[b, c, j0+kk], all 256 channels
        #pragma unroll
        for (int pass = 0; pass < 8; ++pass) {
            int c = pass * 32 + srow;
            const bf16* src = Vbb + (size_t)c * NP + j0 + spart * 8;
            bf16*       dst = &Vs[c][spart * 8];
#if USE_ASYNC_LDS
            __builtin_amdgcn_global_load_async_to_lds_b128((v4i*)src, (v4i*)dst, 0, 0);
#else
            *(bf16x8*)dst = *(const bf16x8*)src;
#endif
        }
        if (j0 + 32 < NP)                           // global_prefetch next K rows
            __builtin_prefetch(kTb + (size_t)(j0 + 32 + ln) * MQ, 0, 1);
#if USE_ASYNC_LDS
        __builtin_amdgcn_s_wait_asynccnt(0);
#endif
        __syncthreads();

        // ---- K B-regs straight from global (2 KB block, L2/WGP$-hot)
        v16bf bk0, bk1;
        {
            const bf16* kp  = kTb + (size_t)(j0 + l16) * MQ + hl * 8;
            const bf16* kp2 = kp + 16 * MQ;
            bk0 = concat8(*(const bf16x8*)kp,  *(const bf16x8*)(kp + 16));
            bk1 = concat8(*(const bf16x8*)kp2, *(const bf16x8*)(kp2 + 16));
        }
        v8f z = {};
        v8f s0 = __builtin_amdgcn_wmma_f32_16x16x32_bf16(false, aq, false, bk0,
                                                         (short)0, z, false, false);
        v8f s1 = __builtin_amdgcn_wmma_f32_16x16x32_bf16(false, aq, false, bk1,
                                                         (short)0, z, false, false);

        // ---- online softmax per row (row r+8*hl lives on one 16-lane half)
        float p0[8], p1[8];
        #pragma unroll
        for (int r = 0; r < 8; ++r) {
            float mx = fmaxf(s0[r], s1[r]);
            #pragma unroll
            for (int msk = 1; msk < 16; msk <<= 1)
                mx = fmaxf(mx, __shfl_xor(mx, msk, 32));
            float nm    = fmaxf(rm[r], mx);
            float alpha = __expf(rm[r] - nm);
            rm[r] = nm;
            p0[r] = __expf(s0[r] - nm);
            p1[r] = __expf(s1[r] - nm);
            float ls = p0[r] + p1[r];
            #pragma unroll
            for (int msk = 1; msk < 16; msk <<= 1)
                ls += __shfl_xor(ls, msk, 32);
            rs[r] = rs[r] * alpha + ls;
            #pragma unroll
            for (int ct = 0; ct < 16; ++ct) acc[ct][r] *= alpha;
        }

        // ---- P: D-layout -> LDS -> A-layout (same wave; DScnt by compiler)
        #pragma unroll
        for (int r = 0; r < 8; ++r) {
            Ps[wv][r + 8 * hl][l16]      = (bf16)p0[r];
            Ps[wv][r + 8 * hl][16 + l16] = (bf16)p1[r];
        }
        v16bf ap;
        {
            const bf16* pp = &Ps[wv][l16][hl * 8];
            ap = concat8(*(const bf16x8*)pp, *(const bf16x8*)(pp + 16));
        }

        // ---- O += P x V^T over 16 channel tiles; B-regs from the shared slab
        #pragma unroll
        for (int ct = 0; ct < 16; ++ct) {
            const bf16* vp = &Vs[ct * 16 + l16][hl * 8];
            v16bf bv = concat8(*(const bf16x8*)vp, *(const bf16x8*)(vp + 16));
            v8f c;
            #pragma unroll
            for (int r = 0; r < 8; ++r) c[r] = acc[ct][r];
            c = __builtin_amdgcn_wmma_f32_16x16x32_bf16(false, ap, false, bv,
                                                        (short)0, c, false, false);
            #pragma unroll
            for (int r = 0; r < 8; ++r) acc[ct][r] = c[r];
        }
        __syncthreads();                            // Vs reused next iteration
    }

    // ---- normalize by rowsum, store O (B, C, NP) f32; rows contiguous over r
    #pragma unroll
    for (int r = 0; r < 8; ++r) rs[r] = 1.0f / rs[r];
    #pragma unroll
    for (int ct = 0; ct < 16; ++ct) {
        int c = ct * 16 + l16;
        float* op = O + ((size_t)b * CDIM + c) * NP + i0 + 8 * hl;
        #pragma unroll
        for (int r = 0; r < 8; ++r) op[r] = acc[ct][r] * rs[r];
    }
}

// ---------------------------------------------------------------------------
// Kernel 3: nearest-neighbor 2x upsample + residual add (pure bandwidth).
// ---------------------------------------------------------------------------
__global__ __launch_bounds__(256)
void upsample_add_kernel(const float* __restrict__ O, const float* __restrict__ Fv,
                         float* __restrict__ out)
{
    size_t idx = (size_t)blockIdx.x * blockDim.x + threadIdx.x;
    const size_t total = (size_t)BSZ * CDIM * HF * WF;
    if (idx >= total) return;
    int w  = (int)(idx & (WF - 1));
    int h  = (int)((idx >> 7) & (HF - 1));
    size_t bc = idx >> 14;                          // b*C + c
    size_t n  = (size_t)(h >> 1) * HP + (w >> 1);
    out[idx] = O[bc * NP + n] + Fv[idx];
}

// ---------------------------------------------------------------------------
// Launch. Workspace layout (bytes):
//   qT  (B,N,32)  bf16 :  2 MB  @ 0
//   kT  (B,N,32)  bf16 :  2 MB  @ 2 MB
//   Vb  (B,256,N) bf16 : 16 MB  @ 4 MB
//   O   (B,256,N) f32  : 32 MB  @ 20 MB            (total ~52 MB)
// ---------------------------------------------------------------------------
extern "C" void kernel_launch(void* const* d_in, const int* in_sizes, int n_in,
                              void* d_out, int out_size, void* d_ws, size_t ws_size,
                              hipStream_t stream)
{
    (void)in_sizes; (void)n_in; (void)out_size; (void)ws_size;
    const float* Fq = (const float*)d_in[0];
    const float* Fk = (const float*)d_in[1];
    const float* Fv = (const float*)d_in[2];
    const float* Wq = (const float*)d_in[3];
    const float* Wk = (const float*)d_in[4];
    const float* Wv = (const float*)d_in[5];
    float* out = (float*)d_out;

    char* ws  = (char*)d_ws;
    bf16*  qT = (bf16*)(ws);
    bf16*  kT = (bf16*)(ws + ((size_t)2 << 20));
    bf16*  Vb = (bf16*)(ws + ((size_t)4 << 20));
    float* O  = (float*)(ws + ((size_t)20 << 20));

    pool_proj_gemm<<<dim3(64, 2, 8),  128, 0, stream>>>(Wq, Fq, qT, MQ, 1);
    pool_proj_gemm<<<dim3(64, 2, 8),  128, 0, stream>>>(Wk, Fk, kT, MQ, 1);
    pool_proj_gemm<<<dim3(64, 16, 8), 128, 0, stream>>>(Wv, Fv, Vb, CDIM, 0);

    flash_attn_kernel<<<dim3(64, 8), 128, 0, stream>>>(qT, kT, Vb, O);

    const size_t total = (size_t)BSZ * CDIM * HF * WF;
    upsample_add_kernel<<<(int)((total + 255) / 256), 256, 0, stream>>>(O, Fv, out);
}
